// MatchNetwork_90726889161130
// MI455X (gfx1250) — matrix operand order repaired and checked by reading, not compiled, not run on hardware
//
#include <hip/hip_runtime.h>

#define BB 32
#define LL 1024
#define DD 256
#define AA 128

typedef __attribute__((ext_vector_type(16))) __bf16 v16bf;
typedef __attribute__((ext_vector_type(8)))  float  v8f;
typedef unsigned short u16;
typedef unsigned int   u32;

union BFrag { v16bf v; uint4 q[2]; u16 u[16]; };

__device__ __forceinline__ u16 f2bf(float f) {
    u32 u = __float_as_uint(f);
    u32 r = (u + 0x7FFFu + ((u >> 16) & 1u)) >> 16;
    return (u16)r;
}

// tanh(x) = 1 - 2/(exp(2x)+1); v_exp_f32 + v_rcp_f32, saturates correctly at +-1
__device__ __forceinline__ float fast_tanh(float x) {
    float e = __builtin_amdgcn_exp2f(x * 2.8853900817779268f); // 2*log2(e)
    return 1.0f - 2.0f * __builtin_amdgcn_rcpf(e + 1.0f);
}

// A-fragment (16x32 bf16, M rows over contiguous K):
//  lane<16:  e0..7 -> K 0..7,   e8..15 -> K 16..23   (row M = lane)
//  lane>=16: e0..7 -> K 8..15,  e8..15 -> K 24..31   (row M = lane-16)
__device__ __forceinline__ v16bf load_frag_a(const u16* base, int ldk, int lane) {
    const u16* p = base + (lane & 15) * ldk + ((lane >> 4) ? 8 : 0);
    BFrag f;
    f.q[0] = *(const uint4*)(p);
    f.q[1] = *(const uint4*)(p + 16);
    return f.v;
}

// B-fragment (32x16 bf16, N cols over contiguous K):
//  lane<16:  e0..7 -> K 0..7,   e8..15 -> K 16..23  (col N = lane)
//  lane>=16: e0..7 -> K 8..15,  e8..15 -> K 24..31  (col N = lane-16)
__device__ __forceinline__ v16bf load_frag_b(const u16* base, int ldk, int lane) {
    const u16* p = base + (lane & 15) * ldk + ((lane >> 4) ? 8 : 0);
    BFrag f;
    f.q[0] = *(const uint4*)(p);
    f.q[1] = *(const uint4*)(p + 16);
    return f.v;
}

#define WMMA_BF16(A, Bf, C) \
    __builtin_amdgcn_wmma_f32_16x16x32_bf16(false, (A), false, (Bf), (short)0, (C), false, false)

// ---------------------------------------------------------------------------
// Cast inputs to bf16; build transposed weight copies for contiguous-K B-frags
// ---------------------------------------------------------------------------
__global__ __launch_bounds__(256) void k_cast(
    const float* __restrict__ f1, const float* __restrict__ f2,
    const float* __restrict__ W,  const float* __restrict__ Wv, const float* __restrict__ Wq,
    u16* __restrict__ f1b, u16* __restrict__ f2b,
    u16* __restrict__ WbT, u16* __restrict__ WvT, u16* __restrict__ WqT)
{
    int i = blockIdx.x * 256 + threadIdx.x;          // grid exactly covers B*L*D
    f1b[i] = f2bf(f1[i]);
    f2b[i] = f2bf(f2[i]);
    if (i < DD * DD) { int k = i / DD, n = i % DD; WbT[n * DD + k] = f2bf(W[i]); }
    if (i < DD * AA) { int d = i / AA, a = i % AA;
                       WvT[a * DD + d] = f2bf(Wv[i]);
                       WqT[a * DD + d] = f2bf(Wq[i]); }
}

// ---------------------------------------------------------------------------
// f1W (bf16) = f1b @ W   : [B*L, D] x [D, D], B operand from WbT[n][k]
// ---------------------------------------------------------------------------
__global__ __launch_bounds__(256) void k_gemm_f1W(
    const u16* __restrict__ Ab_, const u16* __restrict__ WbT, u16* __restrict__ Cb)
{
    const int lane = threadIdx.x & 31, w = threadIdx.x >> 5;
    const int mi = w & 3, wg = w >> 2;               // row tile, col-tile group
    const size_t i0 = (size_t)blockIdx.x * 64;
    const int d0 = blockIdx.y * 64;
    v8f acc[2] = {};
    const u16* Ab = Ab_ + (i0 + 16 * mi) * DD;
    for (int k0 = 0; k0 < DD; k0 += 32) {
        v16bf af = load_frag_a(Ab + k0, DD, lane);
#pragma unroll
        for (int t = 0; t < 2; ++t) {
            v16bf bf = load_frag_b(WbT + (size_t)(d0 + 16 * (wg * 2 + t)) * DD + k0, DD, lane);
            acc[t] = WMMA_BF16(af, bf, acc[t]);
        }
    }
#pragma unroll
    for (int t = 0; t < 2; ++t) {
        int col = d0 + 16 * (wg * 2 + t) + (lane & 15);
#pragma unroll
        for (int r = 0; r < 8; ++r)
            Cb[(i0 + 16 * mi + r + 8 * (lane >> 4)) * DD + col] = f2bf(acc[t][r]);
    }
}

// ---------------------------------------------------------------------------
// Projection GEMM: P = fb @ Wp  ([B*L, A]); writes fp32 [l][a] and bf16 [b][a][l]
// ---------------------------------------------------------------------------
__global__ __launch_bounds__(256) void k_gemm_proj(
    const u16* __restrict__ Ab_, const u16* __restrict__ WT,
    float* __restrict__ Pout, u16* __restrict__ PTb)
{
    const int lane = threadIdx.x & 31, w = threadIdx.x >> 5;
    const int mi = w & 3, wg = w >> 2;
    const size_t row0 = (size_t)blockIdx.x * 64;     // flattened over B*L
    const int a0blk = blockIdx.y * 64;
    const int bIdx = (int)(row0 / LL);
    const int l0 = (int)(row0 % LL);
    v8f acc[2] = {};
    const u16* Ab = Ab_ + (row0 + 16 * mi) * DD;
    for (int k0 = 0; k0 < DD; k0 += 32) {
        v16bf af = load_frag_a(Ab + k0, DD, lane);
#pragma unroll
        for (int t = 0; t < 2; ++t) {
            v16bf bf = load_frag_b(WT + (size_t)(a0blk + 16 * (wg * 2 + t)) * DD + k0, DD, lane);
            acc[t] = WMMA_BF16(af, bf, acc[t]);
        }
    }
#pragma unroll
    for (int t = 0; t < 2; ++t) {
        int acol = a0blk + 16 * (wg * 2 + t) + (lane & 15);
        BFrag pack;
#pragma unroll
        for (int r = 0; r < 8; ++r) {
            float v = acc[t][r];
            Pout[(row0 + 16 * mi + r + 8 * (lane >> 4)) * AA + acol] = v;
            pack.u[r] = f2bf(v);
        }
        int lrow = l0 + 16 * mi + 8 * (lane >> 4);   // 8 consecutive rows -> one b128 store
        *(uint4*)&PTb[(size_t)bIdx * AA * LL + (size_t)acol * LL + lrow] = pack.q[0];
    }
}

// ---------------------------------------------------------------------------
// Fused pass: s[b,i] = ( tanh( P[i,:] + sum_j tanh(X[i,:].Y[j,:]) * Z[j,:] ) ) . wh
// C tiles live only in WMMA accumulators + LDS; Hv/Hq never hit memory.
// The wave's A-strip (16 rows x 256 K of X) is register-resident across all j.
// Hv pass: X=f1W, Y=f2, Z=f2Wq (as ZTb [b][a][l]), P=f1Wv, wh=w_hv -> sv
// Hq pass: X=f2,  Y=f1W, Z=f1Wv,                    P=f2Wq, wh=w_hq -> sq
// ---------------------------------------------------------------------------
__global__ __launch_bounds__(256) void k_pass(
    const u16* __restrict__ Xb, const u16* __restrict__ Yb,
    const u16* __restrict__ ZTb, const float* __restrict__ P,
    const float* __restrict__ wh, float* __restrict__ sOut)
{
    __shared__ u16 cts[64 * 64];                     // tanh(C) tile, bf16, 8 KB
    __shared__ float svl[64];
    const int tid = threadIdx.x;
    const int lane = tid & 31, w = tid >> 5;
    const int mi = w & 3, wg = w >> 2;
    const int b = blockIdx.x >> 4;
    const int i0 = (blockIdx.x & 15) * 64;

    if (tid < 64) svl[tid] = 0.0f;

    // Hoist the A operand strip (reused by all 16 j-tiles): 8 frags = 64 VGPRs
    const u16* Xbase = Xb + (size_t)(b * LL + i0 + 16 * mi) * DD;
    v16bf afrag[DD / 32];
#pragma unroll
    for (int k0 = 0; k0 < DD; k0 += 32)
        afrag[k0 / 32] = load_frag_a(Xbase + k0, DD, lane);

    v8f hacc[4] = {};                                // persistent 64x128 strip (per wave 16x64)

    for (int j0 = 0; j0 < LL; j0 += 64) {
        __syncthreads();                             // cts reuse fence (also orders svl init)
        v8f cacc[2] = {};
#pragma unroll
        for (int k0 = 0; k0 < DD; k0 += 32) {        // C tile: 64x64, K=256
#pragma unroll
            for (int t = 0; t < 2; ++t) {
                const u16* Bb = Yb + (size_t)(b * LL + j0 + 16 * (wg * 2 + t)) * DD + k0;
                v16bf bf = load_frag_b(Bb, DD, lane);
                cacc[t] = WMMA_BF16(afrag[k0 / 32], bf, cacc[t]);
            }
        }
#pragma unroll
        for (int t = 0; t < 2; ++t) {                // tanh -> bf16 -> LDS [row][col]
            int coll = 16 * (wg * 2 + t) + (lane & 15);
#pragma unroll
            for (int r = 0; r < 8; ++r) {
                int rowl = 16 * mi + r + 8 * (lane >> 4);
                cts[rowl * 64 + coll] = f2bf(fast_tanh(cacc[t][r]));
            }
        }
        __syncthreads();
#pragma unroll
        for (int kk = 0; kk < 64; kk += 32) {        // Hacc += Ctile @ Z[j0:j0+64, :]
            v16bf af = load_frag_a(&cts[(16 * mi) * 64 + kk], 64, lane);
#pragma unroll
            for (int t = 0; t < 4; ++t) {
                int a0 = 16 * (wg * 4 + t);
                const u16* Bb = ZTb + (size_t)b * AA * LL + (size_t)a0 * LL + j0 + kk;
                v16bf bf = load_frag_b(Bb, LL, lane);
                hacc[t] = WMMA_BF16(af, bf, hacc[t]);
            }
        }
    }
    // epilogue: h = tanh(P + hacc); s[row] = sum_a h*wh[a]
#pragma unroll
    for (int t = 0; t < 4; ++t) {
        int acol = 16 * (wg * 4 + t) + (lane & 15);
        float whv = wh[acol];
#pragma unroll
        for (int r = 0; r < 8; ++r) {
            int li = i0 + 16 * mi + r + 8 * (lane >> 4);
            float h = fast_tanh(P[(size_t)(b * LL + li) * AA + acol] + hacc[t][r]);
            atomicAdd(&svl[16 * mi + r + 8 * (lane >> 4)], h * whv);
        }
    }
    __syncthreads();
    if (tid < 64) sOut[(size_t)b * LL + i0 + tid] = svl[tid];
}

// ---------------------------------------------------------------------------
// Masked softmax (memory_efficient=False semantics) + attention pooling
// ---------------------------------------------------------------------------
__global__ __launch_bounds__(256) void k_softmax_pool(
    const float* __restrict__ s, const int* __restrict__ mask,
    const float* __restrict__ f, float* __restrict__ outp)
{
    const int b = blockIdx.x, tid = threadIdx.x;
    __shared__ float p[LL];
    __shared__ float red[256];

    float lmax = -3.4e38f;
    for (int l = tid; l < LL; l += 256) {
        float m = (float)mask[b * LL + l];
        float x = s[b * LL + l] * m;                 // softmax over (vector * mask)
        p[l] = x;
        lmax = fmaxf(lmax, x);
    }
    red[tid] = lmax; __syncthreads();
    for (int o = 128; o > 0; o >>= 1) { if (tid < o) red[tid] = fmaxf(red[tid], red[tid + o]); __syncthreads(); }
    float mx = red[0]; __syncthreads();

    float lsum = 0.0f;
    for (int l = tid; l < LL; l += 256) {
        float e = __builtin_amdgcn_exp2f((p[l] - mx) * 1.4426950408889634f);
        p[l] = e; lsum += e;
    }
    red[tid] = lsum; __syncthreads();
    for (int o = 128; o > 0; o >>= 1) { if (tid < o) red[tid] += red[tid + o]; __syncthreads(); }
    float inv1 = 1.0f / red[0]; __syncthreads();

    float lsum2 = 0.0f;
    for (int l = tid; l < LL; l += 256) {            // re-mask, then renormalize
        float m = (float)mask[b * LL + l];
        float q = p[l] * inv1 * m;
        p[l] = q; lsum2 += q;
    }
    red[tid] = lsum2; __syncthreads();
    for (int o = 128; o > 0; o >>= 1) { if (tid < o) red[tid] += red[tid + o]; __syncthreads(); }
    float inv2 = 1.0f / (red[0] + 1e-13f); __syncthreads();
    for (int l = tid; l < LL; l += 256) p[l] *= inv2;
    __syncthreads();

    float acc = 0.0f;                                // pooled[d] = sum_l p[l]*f[b,l,d], d = tid
    for (int l = 0; l < LL; ++l)
        acc += p[l] * f[((size_t)b * LL + l) * DD + tid];
    outp[(size_t)b * DD + tid] = acc;
}

// ---------------------------------------------------------------------------
extern "C" void kernel_launch(void* const* d_in, const int* in_sizes, int n_in,
                              void* d_out, int out_size, void* d_ws, size_t ws_size,
                              hipStream_t stream)
{
    (void)in_sizes; (void)n_in; (void)out_size; (void)ws_size;
    const float* f1   = (const float*)d_in[0];
    const float* f2   = (const float*)d_in[1];
    const int*   m1   = (const int*)  d_in[2];
    const int*   m2   = (const int*)  d_in[3];
    const float* W    = (const float*)d_in[4];
    const float* Wv   = (const float*)d_in[5];
    const float* Wq   = (const float*)d_in[6];
    const float* whv  = (const float*)d_in[7];
    const float* whq  = (const float*)d_in[8];
    float* out = (float*)d_out;

    const size_t NBL = (size_t)BB * LL;
    u16*   f1b    = (u16*)d_ws;                      // 16 MB
    u16*   f2b    = f1b + NBL * DD;                  // 16 MB
    u16*   f1Wb   = f2b + NBL * DD;                  // 16 MB
    float* f1Wv   = (float*)(f1Wb + NBL * DD);       // 16 MB
    float* f2Wq   = f1Wv + NBL * AA;                 // 16 MB
    u16*   f1WvTb = (u16*)(f2Wq + NBL * AA);         // 8 MB  [b][a][l]
    u16*   f2WqTb = f1WvTb + NBL * AA;               // 8 MB  [b][a][l]
    float* sv     = (float*)(f2WqTb + NBL * AA);     // 128 KB
    float* sq     = sv + NBL;                        // 128 KB
    u16*   WbT    = (u16*)(sq + NBL);                // 128 KB
    u16*   WvT    = WbT + DD * DD;                   // 64 KB
    u16*   WqT    = WvT + DD * AA;                   // 64 KB

    k_cast<<<(BB * LL * DD) / 256, 256, 0, stream>>>(f1, f2, W, Wv, Wq, f1b, f2b, WbT, WvT, WqT);
    k_gemm_f1W<<<dim3(NBL / 64, DD / 64), 256, 0, stream>>>(f1b, WbT, f1Wb);
    k_gemm_proj<<<dim3(NBL / 64, AA / 64), 256, 0, stream>>>(f1b, WvT, f1Wv, f1WvTb);
    k_gemm_proj<<<dim3(NBL / 64, AA / 64), 256, 0, stream>>>(f2b, WqT, f2Wq, f2WqTb);
    k_pass<<<BB * (LL / 64), 256, 0, stream>>>(f1Wb, f2b, f2WqTb, f1Wv, whv, sv);
    k_pass<<<BB * (LL / 64), 256, 0, stream>>>(f2b, f1Wb, f1WvTb, f2Wq, whq, sq);
    k_softmax_pool<<<BB, 256, 0, stream>>>(sv, m1, f1, out);
    k_softmax_pool<<<BB, 256, 0, stream>>>(sq, m2, f2, out + BB * DD);
}
